// KimiMoE_4879082848961
// MI455X (gfx1250) — compile-verified, matrix operand order checked by reading, not run on hardware
//
#include <hip/hip_runtime.h>
#include <hip/hip_bf16.h>
#include <stdint.h>

// ---------------------------------------------------------------------------
// MoE (router + shared expert + top-2 sparse experts), bf16 WMMA on gfx1250.
// Weights pre-packed (fp32 -> bf16, WMMA B-fragment lane layout) once per call;
// GEMM hot loops are pure load+wmma with 2x B-fragment reuse per wave.
// ---------------------------------------------------------------------------

typedef __attribute__((ext_vector_type(16))) __bf16 v16bf;
typedef __attribute__((ext_vector_type(8)))  float  v8f;

static constexpr int Bc = 2, Sc = 2048, Dc = 1024, Ec = 8, Fc = 2048;
static constexpr int Tc = Bc * Sc;                   // 4096 tokens
static constexpr int BMt = 64, BN = 128;             // block tile 64 x 128
static constexpr int KCH = 512;                      // K chunk staged in LDS
static constexpr int SLOT_CAP = Tc * 2 + Ec * BMt;   // padded routed slots (8704)
static constexpr int PSTR = 40;                      // pack-LDS row stride (bf16)
static constexpr int AST  = KCH + 8;                 // A-chunk LDS row stride (520 bf16, 16B-aligned rows)

__device__ __forceinline__ unsigned short f2bf(float f) {
  unsigned int u = __float_as_uint(f);
  u += 0x7FFFu + ((u >> 16) & 1u);                   // round-to-nearest-even
  return (unsigned short)(u >> 16);
}

union FragBF { uint4 q[2]; v16bf v; };

__device__ __forceinline__ v8f wmma_bf16(const v16bf& a, const v16bf& b, const v8f& c) {
  return __builtin_amdgcn_wmma_f32_16x16x32_bf16(false, a, false, b, (short)0, c, false, false);
}

// A fragment from LDS chunk (row-major over K, matches 16-bit WMMA lane layout).
__device__ __forceinline__ v16bf ldsFragA(const unsigned short* As, int row, int kOfs, int half) {
  FragBF f;
  const unsigned short* p = As + row * AST + kOfs + half * 8;
  f.q[0] = *(const uint4*)(p);
  f.q[1] = *(const uint4*)(p + 16);
  return f.v;
}

// B fragment from packed global weights: per-lane 32 contiguous bytes.
__device__ __forceinline__ v16bf gblFragB(const unsigned short* p) {
  FragBF f;
  f.q[0] = *(const uint4*)(p);
  f.q[1] = *(const uint4*)(p + 8);
  return f.v;
}

// ---------------------------------------------------------------------------
// Pack W [K,N] fp32 row-major -> bf16 fragment layout:
//   P[(((kt*NT16 + n16)*32 + lane)*16 .. +15]  (32B per lane per 32x16 tile)
// grid: (K/32, N/128, nMat)
// ---------------------------------------------------------------------------
__global__ __launch_bounds__(256) void moe_pack_w(const float* __restrict__ W,
                                                  unsigned short* __restrict__ P,
                                                  int K, int N) {
  __shared__ unsigned short BT[BN * PSTR];
  const size_t matOfs = (size_t)blockIdx.z * K * N;
  const float* Wm = W + matOfs;
  unsigned short* Pm = P + matOfs;
  const int k0 = blockIdx.x * 32;
  const int n0 = blockIdx.y * 128;
  const int kB = threadIdx.x >> 3;
  const int nB = (threadIdx.x & 7) << 4;
  const float* p = Wm + (size_t)(k0 + kB) * N + n0 + nB;
#pragma unroll
  for (int j = 0; j < 16; ++j) BT[(nB + j) * PSTR + kB] = f2bf(p[j]);
  __syncthreads();
  const int lane = threadIdx.x & 31;
  const int wave = threadIdx.x >> 5;        // n16 sub-tile 0..7
  const int half = (lane >> 4) & 1;
  FragBF f;
  const unsigned short* q = BT + (wave * 16 + (lane & 15)) * PSTR + half * 8;
  f.q[0] = *(const uint4*)(q);
  f.q[1] = *(const uint4*)(q + 16);
  const int NT16 = N >> 4;
  unsigned short* dst = Pm + (((size_t)blockIdx.x * NT16 + (n0 >> 4) + wave) * 32 + lane) * 16;
  *(uint4*)(dst)     = f.q[0];
  *(uint4*)(dst + 8) = f.q[1];
}

// ---------------------------------------------------------------------------
// X (fp32) -> bf16 copy
// ---------------------------------------------------------------------------
__global__ __launch_bounds__(256) void moe_cvt_bf16(const float* __restrict__ X,
                                                    unsigned short* __restrict__ Xbf) {
  size_t i = ((size_t)blockIdx.x * 256 + threadIdx.x) * 4;
  float4 v = *(const float4*)(X + i);
  unsigned int lo = (unsigned int)f2bf(v.x) | ((unsigned int)f2bf(v.y) << 16);
  unsigned int hi = (unsigned int)f2bf(v.z) | ((unsigned int)f2bf(v.w) << 16);
  *(uint2*)(Xbf + i) = make_uint2(lo, hi);
}

// ---------------------------------------------------------------------------
// Router: one wave per token; softmax -> top2 (first-max ties) -> softmax of
// the two probabilities (matches reference exactly).
// ---------------------------------------------------------------------------
__global__ __launch_bounds__(256) void moe_router(const float* __restrict__ X,
                                                  const float* __restrict__ Wr,
                                                  int* __restrict__ topkIdx,
                                                  float* __restrict__ topkW) {
  const int lane = threadIdx.x & 31;
  const int wave = threadIdx.x >> 5;
  const int t = blockIdx.x * 8 + wave;
  float acc[Ec];
#pragma unroll
  for (int e = 0; e < Ec; ++e) acc[e] = 0.0f;
  const float* xp = X + (size_t)t * Dc;
  for (int d = lane; d < Dc; d += 32) {
    float x = xp[d];
    const float* wr = Wr + (size_t)d * Ec;
#pragma unroll
    for (int e = 0; e < Ec; ++e) acc[e] += x * wr[e];
  }
#pragma unroll
  for (int off = 16; off >= 1; off >>= 1) {
#pragma unroll
    for (int e = 0; e < Ec; ++e) acc[e] += __shfl_xor(acc[e], off, 32);
  }
  if (lane == 0) {
    float m = acc[0];
#pragma unroll
    for (int e = 1; e < Ec; ++e) m = fmaxf(m, acc[e]);
    float p[Ec], s = 0.0f;
#pragma unroll
    for (int e = 0; e < Ec; ++e) { p[e] = __expf(acc[e] - m); s += p[e]; }
    float inv = 1.0f / s;
#pragma unroll
    for (int e = 0; e < Ec; ++e) p[e] *= inv;
    int e0 = 0;
#pragma unroll
    for (int e = 1; e < Ec; ++e) if (p[e] > p[e0]) e0 = e;
    int e1 = (e0 == 0) ? 1 : 0;
#pragma unroll
    for (int e = 0; e < Ec; ++e) if (e != e0 && p[e] > p[e1]) e1 = e;
    float a = p[e0], b = p[e1];
    float mm = fmaxf(a, b);
    float ea = __expf(a - mm), eb = __expf(b - mm);
    float di = 1.0f / (ea + eb);
    topkIdx[t * 2 + 0] = e0;    topkIdx[t * 2 + 1] = e1;
    topkW[t * 2 + 0] = ea * di; topkW[t * 2 + 1] = eb * di;
  }
}

// ---------------------------------------------------------------------------
// Deterministic per-expert compaction (single block, block-wide scan).
// Slot lists padded to BMt so every M-tile belongs to exactly one expert.
// ---------------------------------------------------------------------------
__global__ __launch_bounds__(256) void moe_route_build(const int* __restrict__ topkIdx,
                                                       const float* __restrict__ topkW,
                                                       int* __restrict__ slotIdx,
                                                       int* __restrict__ slotRank,
                                                       float* __restrict__ slotW,
                                                       int* __restrict__ offA) {
  __shared__ int scan[256];
  const int tid = threadIdx.x;
  int gstart = 0;
  for (int e = 0; e < Ec; ++e) {
    int cnt = 0;
    for (int c = 0; c < Tc / 256; ++c) {
      int t = c * 256 + tid;
      int e0 = topkIdx[2 * t], e1 = topkIdx[2 * t + 1];
      int match = (e0 == e || e1 == e) ? 1 : 0;
      int rk = (e0 == e) ? 0 : 1;
      scan[tid] = match;
      __syncthreads();
      for (int off = 1; off < 256; off <<= 1) {
        int v = (tid >= off) ? scan[tid - off] : 0;
        __syncthreads();
        scan[tid] += v;
        __syncthreads();
      }
      int pos = scan[tid] - match;
      int tot = scan[255];
      __syncthreads();
      if (match) {
        int s = gstart + cnt + pos;
        slotIdx[s] = t;
        slotRank[s] = rk;
        slotW[s] = topkW[2 * t + rk];
      }
      cnt += tot;
    }
    int padded = (cnt + BMt - 1) & ~(BMt - 1);
    for (int s = cnt + tid; s < padded; s += 256) {
      slotIdx[gstart + s] = -1; slotRank[gstart + s] = 0; slotW[gstart + s] = 0.0f;
    }
    if (tid == 0) offA[e] = gstart;
    gstart += padded;
    __syncthreads();
  }
  if (tid == 0) offA[Ec] = gstart;
}

// ---------------------------------------------------------------------------
// GEMM1: H = silu(A @ Wg) * (A @ Wu). Block tile 64x128; 8 waves (2x4); each
// wave 32M x 32N -> per K-step: 4 ds_load_b128 + 8 global_load_b128 + 8 WMMA.
// A staged in 512-wide K chunks (66 KB LDS -> 4 blocks/WGP).
// ---------------------------------------------------------------------------
__global__ __launch_bounds__(256) void moe_gemm1(const unsigned short* __restrict__ Abf,
                                                 const unsigned short* __restrict__ WgP,
                                                 const unsigned short* __restrict__ WuP,
                                                 unsigned short* __restrict__ H,
                                                 const int* __restrict__ slotIdx,
                                                 const int* __restrict__ offA) {
  constexpr int NT16 = Fc / 16;                      // 128
  __shared__ unsigned short As[BMt * AST];           // 64*520*2 = 66560 B
  __shared__ int sTok[BMt];

  const int mBase = blockIdx.x * BMt;
  const int nBase = blockIdx.y * BN;

  const unsigned short* Wg = WgP;
  const unsigned short* Wu = WuP;
  if (slotIdx) {
    if (mBase >= offA[Ec]) return;
    int e = 0;
    while (e < Ec - 1 && mBase >= offA[e + 1]) e++;
    const size_t wofs = (size_t)e * Dc * Fc;
    Wg += wofs; Wu += wofs;
  }
  if (threadIdx.x < BMt) {
    int s = mBase + threadIdx.x;
    sTok[threadIdx.x] = slotIdx ? slotIdx[s] : s;
  }
  __syncthreads();

  const int lane = threadIdx.x & 31;
  const int wave = threadIdx.x >> 5;
  const int wm = wave >> 2, wn = wave & 3;
  const int half = (lane >> 4) & 1;
  const int ml = lane & 15;
  const int rS = threadIdx.x >> 2;                   // 64 rows, 4 threads each
  const int cb = (threadIdx.x & 3) << 7;             // 128 bf16 per thread
  const int tokS = sTok[rS];

  const int n16a = (nBase >> 4) + wn * 2;
  const unsigned short* pg0 = Wg + ((size_t)(n16a + 0) * 32 + lane) * 16;
  const unsigned short* pg1 = Wg + ((size_t)(n16a + 1) * 32 + lane) * 16;
  const unsigned short* pu0 = Wu + ((size_t)(n16a + 0) * 32 + lane) * 16;
  const unsigned short* pu1 = Wu + ((size_t)(n16a + 1) * 32 + lane) * 16;
  constexpr size_t KSTEP = (size_t)NT16 * 32 * 16;   // packed elems per kt

  const v8f z8 = {0,0,0,0,0,0,0,0};
  v8f accG[2][2] = {{z8, z8}, {z8, z8}};
  v8f accU[2][2] = {{z8, z8}, {z8, z8}};

  for (int c = 0; c < Dc / KCH; ++c) {
    __syncthreads();
    const unsigned short* src = Abf + (size_t)(tokS < 0 ? 0 : tokS) * Dc + c * KCH + cb;
#pragma unroll
    for (int j = 0; j < 16; ++j) {
      uint4 v = make_uint4(0u, 0u, 0u, 0u);
      if (tokS >= 0) v = *(const uint4*)(src + j * 8);
      *(uint4*)(&As[rS * AST + cb + j * 8]) = v;
    }
    __syncthreads();

    for (int kt = 0; kt < KCH / 32; ++kt) {
      v16bf fa0 = ldsFragA(As, wm * 32 + ml,      kt * 32, half);
      v16bf fa1 = ldsFragA(As, wm * 32 + 16 + ml, kt * 32, half);
      v16bf g0 = gblFragB(pg0);
      v16bf g1 = gblFragB(pg1);
      v16bf u0 = gblFragB(pu0);
      v16bf u1 = gblFragB(pu1);
      __builtin_prefetch(pg0 + KSTEP, 0, 3);         // near-cache prefetch
      __builtin_prefetch(pu0 + KSTEP, 0, 3);
      accG[0][0] = wmma_bf16(fa0, g0, accG[0][0]);
      accG[0][1] = wmma_bf16(fa0, g1, accG[0][1]);
      accU[0][0] = wmma_bf16(fa0, u0, accU[0][0]);
      accU[0][1] = wmma_bf16(fa0, u1, accU[0][1]);
      accG[1][0] = wmma_bf16(fa1, g0, accG[1][0]);
      accG[1][1] = wmma_bf16(fa1, g1, accG[1][1]);
      accU[1][0] = wmma_bf16(fa1, u0, accU[1][0]);
      accU[1][1] = wmma_bf16(fa1, u1, accU[1][1]);
      pg0 += KSTEP; pg1 += KSTEP; pu0 += KSTEP; pu1 += KSTEP;
    }
  }

  const int mb = half * 8;
#pragma unroll
  for (int mf = 0; mf < 2; ++mf) {
#pragma unroll
    for (int fi = 0; fi < 2; ++fi) {
      v8f g = accG[mf][fi];
      v8f u = accU[mf][fi];
      const int col = nBase + wn * 32 + fi * 16 + ml;
#pragma unroll
      for (int i = 0; i < 8; ++i) {
        float gv = g[i];
        float hv = (gv / (1.0f + __expf(-gv))) * u[i];   // silu(g) * u
        int row = mBase + wm * 32 + mf * 16 + mb + i;
        H[(size_t)row * Fc + col] = f2bf(hv);
      }
    }
  }
}

// ---------------------------------------------------------------------------
// GEMM2: Out = H @ Wd. K = F = 2048 staged as four 512 LDS chunks.
// Shared mode writes d_out; expert mode scales by routing weight and scatters
// into rank-separated Y (no fp atomics -> bit-deterministic).
// ---------------------------------------------------------------------------
__global__ __launch_bounds__(256) void moe_gemm2(const unsigned short* __restrict__ Hbf,
                                                 const unsigned short* __restrict__ WdP,
                                                 float* __restrict__ Out,
                                                 float* __restrict__ Y,
                                                 const int* __restrict__ slotIdx,
                                                 const int* __restrict__ slotRank,
                                                 const float* __restrict__ slotW,
                                                 const int* __restrict__ offA) {
  constexpr int NT16 = Dc / 16;                      // 64
  __shared__ unsigned short As[BMt * AST];
  __shared__ int sTok[BMt];
  __shared__ int sRk[BMt];
  __shared__ float sW[BMt];

  const int mBase = blockIdx.x * BMt;
  const int nBase = blockIdx.y * BN;

  const unsigned short* Wd = WdP;
  if (slotIdx) {
    if (mBase >= offA[Ec]) return;
    int e = 0;
    while (e < Ec - 1 && mBase >= offA[e + 1]) e++;
    Wd += (size_t)e * Fc * Dc;
  }
  if (threadIdx.x < BMt) {
    int s = mBase + threadIdx.x;
    if (slotIdx) { sTok[threadIdx.x] = slotIdx[s]; sRk[threadIdx.x] = slotRank[s]; sW[threadIdx.x] = slotW[s]; }
    else         { sTok[threadIdx.x] = s;          sRk[threadIdx.x] = 0;           sW[threadIdx.x] = 1.0f; }
  }

  const int lane = threadIdx.x & 31;
  const int wave = threadIdx.x >> 5;
  const int wm = wave >> 2, wn = wave & 3;
  const int half = (lane >> 4) & 1;
  const int ml = lane & 15;
  const int rS = threadIdx.x >> 2;
  const int cb = (threadIdx.x & 3) << 7;

  const int n16a = (nBase >> 4) + wn * 2;
  constexpr size_t KSTEP = (size_t)NT16 * 32 * 16;
  const unsigned short* pb0 = Wd + ((size_t)(n16a + 0) * 32 + lane) * 16;
  const unsigned short* pb1 = Wd + ((size_t)(n16a + 1) * 32 + lane) * 16;

  const v8f z8 = {0,0,0,0,0,0,0,0};
  v8f acc[2][2] = {{z8, z8}, {z8, z8}};

  for (int c = 0; c < Fc / KCH; ++c) {
    __syncthreads();
    // A rows are slots/tokens directly (He pad rows are valid).
    const unsigned short* src = Hbf + (size_t)(mBase + rS) * Fc + c * KCH + cb;
#pragma unroll
    for (int j = 0; j < 16; ++j)
      *(uint4*)(&As[rS * AST + cb + j * 8]) = *(const uint4*)(src + j * 8);
    __syncthreads();

    for (int kt = 0; kt < KCH / 32; ++kt) {
      v16bf fa0 = ldsFragA(As, wm * 32 + ml,      kt * 32, half);
      v16bf fa1 = ldsFragA(As, wm * 32 + 16 + ml, kt * 32, half);
      v16bf b0 = gblFragB(pb0);
      v16bf b1 = gblFragB(pb1);
      __builtin_prefetch(pb0 + KSTEP, 0, 3);
      acc[0][0] = wmma_bf16(fa0, b0, acc[0][0]);
      acc[0][1] = wmma_bf16(fa0, b1, acc[0][1]);
      acc[1][0] = wmma_bf16(fa1, b0, acc[1][0]);
      acc[1][1] = wmma_bf16(fa1, b1, acc[1][1]);
      pb0 += KSTEP; pb1 += KSTEP;
    }
  }

  const int mb = half * 8;
#pragma unroll
  for (int mf = 0; mf < 2; ++mf) {
#pragma unroll
    for (int fi = 0; fi < 2; ++fi) {
      v8f a = acc[mf][fi];
      const int col = nBase + wn * 32 + fi * 16 + ml;
#pragma unroll
      for (int i = 0; i < 8; ++i) {
        int rl = wm * 32 + mf * 16 + mb + i;
        int tok = sTok[rl];
        if (tok < 0) continue;
        float val = a[i];
        if (slotIdx) {
          Y[(size_t)sRk[rl] * Tc * Dc + (size_t)tok * Dc + col] = sW[rl] * val;
        } else {
          Out[(size_t)tok * Dc + col] = val;
        }
      }
    }
  }
}

// ---------------------------------------------------------------------------
// out += Y0 + Y1
// ---------------------------------------------------------------------------
__global__ __launch_bounds__(256) void moe_combine(float* __restrict__ Out,
                                                   const float* __restrict__ Y) {
  size_t i = ((size_t)blockIdx.x * 256 + threadIdx.x) * 4;
  float4 o  = *(float4*)(Out + i);
  float4 y0 = *(const float4*)(Y + i);
  float4 y1 = *(const float4*)(Y + (size_t)Tc * Dc + i);
  o.x += y0.x + y1.x;  o.y += y0.y + y1.y;
  o.z += y0.z + y1.z;  o.w += y0.w + y1.w;
  *(float4*)(Out + i) = o;
}

// ---------------------------------------------------------------------------
extern "C" void kernel_launch(void* const* d_in, const int* in_sizes, int n_in,
                              void* d_out, int out_size, void* d_ws, size_t ws_size,
                              hipStream_t stream) {
  (void)in_sizes; (void)n_in; (void)out_size; (void)ws_size;
  const float* X    = (const float*)d_in[0];   // [B,S,D]
  const float* Wr   = (const float*)d_in[1];   // [D,E]
  const float* Wg_s = (const float*)d_in[2];   // [D,F]
  const float* Wu_s = (const float*)d_in[3];   // [D,F]
  const float* Wd_s = (const float*)d_in[4];   // [F,D]
  const float* Wg   = (const float*)d_in[5];   // [E,D,F]
  const float* Wu   = (const float*)d_in[6];   // [E,D,F]
  const float* Wd   = (const float*)d_in[7];   // [E,F,D]
  float* out = (float*)d_out;                  // [B,S,D] fp32

  char* ws = (char*)d_ws;
  size_t o = 0;
  auto take = [&](size_t bytes) -> char* {
    o = (o + 255) & ~(size_t)255;
    char* p = ws + o;
    o += bytes;
    return p;
  };
  int*            topkIdx  = (int*)take((size_t)Tc * 2 * sizeof(int));
  float*          topkW    = (float*)take((size_t)Tc * 2 * sizeof(float));
  int*            slotIdx  = (int*)take((size_t)SLOT_CAP * sizeof(int));
  int*            slotRank = (int*)take((size_t)SLOT_CAP * sizeof(int));
  float*          slotW    = (float*)take((size_t)SLOT_CAP * sizeof(float));
  int*            offA     = (int*)take(16 * sizeof(int));
  unsigned short* Xbf      = (unsigned short*)take((size_t)Tc * Dc * 2);
  unsigned short* Hs       = (unsigned short*)take((size_t)Tc * Fc * 2);
  unsigned short* He       = (unsigned short*)take((size_t)SLOT_CAP * Fc * 2);
  float*          Y        = (float*)take((size_t)2 * Tc * Dc * sizeof(float));
  unsigned short* WgsP     = (unsigned short*)take((size_t)Dc * Fc * 2);
  unsigned short* WusP     = (unsigned short*)take((size_t)Dc * Fc * 2);
  unsigned short* WdsP     = (unsigned short*)take((size_t)Fc * Dc * 2);
  unsigned short* WgP      = (unsigned short*)take((size_t)Ec * Dc * Fc * 2);
  unsigned short* WuP      = (unsigned short*)take((size_t)Ec * Dc * Fc * 2);
  unsigned short* WdP      = (unsigned short*)take((size_t)Ec * Fc * Dc * 2);

  // 0) pack weights fp32 -> bf16 fragment layout (per call; graph-safe)
  dim3 pkDF(Dc / 32, Fc / 128, 1);
  dim3 pkFD(Fc / 32, Dc / 128, 1);
  dim3 pkDFe(Dc / 32, Fc / 128, Ec);
  dim3 pkFDe(Fc / 32, Dc / 128, Ec);
  moe_pack_w<<<pkDF, 256, 0, stream>>>(Wg_s, WgsP, Dc, Fc);
  moe_pack_w<<<pkDF, 256, 0, stream>>>(Wu_s, WusP, Dc, Fc);
  moe_pack_w<<<pkFD, 256, 0, stream>>>(Wd_s, WdsP, Fc, Dc);
  moe_pack_w<<<pkDFe, 256, 0, stream>>>(Wg, WgP, Dc, Fc);
  moe_pack_w<<<pkDFe, 256, 0, stream>>>(Wu, WuP, Dc, Fc);
  moe_pack_w<<<pkFDe, 256, 0, stream>>>(Wd, WdP, Fc, Dc);
  // 1) X -> bf16
  moe_cvt_bf16<<<(Tc * Dc) / 1024, 256, 0, stream>>>(X, Xbf);
  // 2) router
  moe_router<<<Tc / 8, 256, 0, stream>>>(X, Wr, topkIdx, topkW);
  // 3) deterministic per-expert compaction
  moe_route_build<<<1, 256, 0, stream>>>(topkIdx, topkW, slotIdx, slotRank, slotW, offA);
  // 4) shared expert GEMM1 (gate+up fused, SiLU epilogue)
  dim3 g1s(Tc / BMt, Fc / BN);
  moe_gemm1<<<g1s, 256, 0, stream>>>(Xbf, WgsP, WusP, Hs, nullptr, nullptr);
  // 5) routed experts GEMM1
  dim3 g1e(SLOT_CAP / BMt, Fc / BN);
  moe_gemm1<<<g1e, 256, 0, stream>>>(Xbf, WgP, WuP, He, slotIdx, offA);
  // 6) shared expert GEMM2 -> d_out
  dim3 g2s(Tc / BMt, Dc / BN);
  moe_gemm2<<<g2s, 256, 0, stream>>>(Hs, WdsP, out, nullptr, nullptr, nullptr, nullptr, nullptr);
  // 7) routed experts GEMM2 -> Y[rank]
  dim3 g2e(SLOT_CAP / BMt, Dc / BN);
  moe_gemm2<<<g2e, 256, 0, stream>>>(He, WdP, nullptr, Y, slotIdx, slotRank, slotW, offA);
  // 8) out += Y0 + Y1
  moe_combine<<<(Tc * Dc) / 1024, 256, 0, stream>>>(out, Y);
}
